// CNN_84241488544497
// MI455X (gfx1250) — matrix operand order repaired and checked by reading, not compiled, not run on hardware
//
#include <hip/hip_runtime.h>

typedef __attribute__((ext_vector_type(16))) _Float16 v16h;
typedef __attribute__((ext_vector_type(8)))  _Float16 v8h;
typedef __attribute__((ext_vector_type(8)))  float    v8f;
typedef __attribute__((ext_vector_type(4)))  float    v4f;

#define B_SEQ 512
#define L_SEQ 512
#define A_AA  20
#define C_CH  8
#define ROWS  5          // amino rows per workgroup -> 4 WGs per sequence
#define WGS   256        // 8 wave32 waves

// Activation LDS layout: [row][pos_slot][ch], pos_slot = pos+1 (zero pad slot
// on each side). One slot = 8 fp16 channels = 16B = one ds_load/store_b128.
//
// conv(8->8,k=3,pad1)+AvgPool(2) folded into ONE GEMM per 16 pooled outputs:
//   pool[o][q] = sum_{tau=0..3, c} W2[o][tau*8+c] * in[c][2q-1+tau]
//   W2 taps = {w0/2, (w0+w1)/2, (w1+w2)/2, w2/2}   -> K = 32, fully used.
// WMMA K mapping: k = tau*8 + c.

template <int LIN, int LIDX, bool LAST>
__device__ __forceinline__ void conv_pool_layer(const v8h* __restrict__ inb,
                                                v8h* __restrict__ outb,
                                                const v8h* __restrict__ sW,
                                                float* __restrict__ outg,
                                                int lane, int wave,
                                                int outRowBase) {
  constexpr int SIN  = LIN + 2;
  constexpr int LOUT = LIN >> 1;
  constexpr int SOUT = LOUT + 2;
  constexpr int NT   = (LOUT >= 16) ? (LOUT >> 4) : 1;
  constexpr int TSH  = __builtin_ctz(NT);
  constexpr int NLIM = (LOUT >= 16) ? 16 : LOUT;   // lanes that store

  const int m  = lane & 15;
  const int hi = (lane >> 4) & 1;                  // 0: k=0..15, 1: k=16..31

  // A fragment (pool-folded weights): two 16B LDS loads, zeros for m>=8.
  const v8h* wb = sW + (LIDX * 16 + m) * 4;        // [tau] slots of 8 ch
  v8h alo = wb[2 * hi];                            // tau = 2*hi   -> j=0..7
  v8h ahi = wb[2 * hi + 1];                        // tau = 2*hi+1 -> j=8..15
  v16h afrag = __builtin_shufflevector(alo, ahi, 0, 1, 2, 3, 4, 5, 6, 7, 8, 9,
                                       10, 11, 12, 13, 14, 15);

  for (int u = wave; u < ROWS * NT; u += 8) {      // wave-uniform trip count
    const int row = u >> TSH;
    const int tb  = (u - (row << TSH)) << 4;
    const int q   = tb + (lane & 15);              // pooled output position
    const int qc  = (LOUT >= 16) ? q : (q < LOUT ? q : LOUT - 1);  // clamp

    // B fragment: two contiguous unconditional 16B LDS loads.
    // lanes 0-15 : slots 2q,2q+1   (pos 2q-1, 2q   ; tau 0,1)
    // lanes 16-31: slots 2q+2,2q+3 (pos 2q+1, 2q+2 ; tau 2,3)
    const v8h* base = inb + row * SIN + (qc << 1) + (hi << 1);
    v8h blo = base[0];
    v8h bhi = base[1];
    v16h bfrag = __builtin_shufflevector(blo, bhi, 0, 1, 2, 3, 4, 5, 6, 7, 8,
                                         9, 10, 11, 12, 13, 14, 15);

    v8f acc = {};
    acc = __builtin_amdgcn_wmma_f32_16x16x32_f16(false, afrag, false, bfrag,
                                                 (short)0, acc, false, false);

    // Lane n's accumulator column IS the pooled output (8 channels).
    if (lane < NLIM) {
      if constexpr (LAST) {
        float* dst = outg + (size_t)(outRowBase + row) * C_CH;
#pragma unroll
        for (int r = 0; r < 8; ++r) dst[r] = acc[r];
      } else {
        v8h pk;
#pragma unroll
        for (int r = 0; r < 8; ++r) pk[r] = (_Float16)acc[r];
        outb[row * SOUT + q + 1] = pk;             // single ds_store_b128
      }
    }
  }

  if constexpr (!LAST) {
    const int tid = (wave << 5) | lane;
    if (tid < ROWS * 2) {                          // zero the two pad slots
      int r = tid >> 1;
      outb[r * SOUT + ((tid & 1) ? (LOUT + 1) : 0)] = (v8h){};
    }
  }
  __syncthreads();
}

__global__ __launch_bounds__(WGS) void protcnn_fused(
    const float* __restrict__ x,    // (B, L, 20) one-hot
    const float* __restrict__ lpm,  // (20,20)
    const float* __restrict__ pm,   // (20,20)
    const float* __restrict__ wf,   // (8,1,3)
    const float* __restrict__ wr,   // (8,8,8,3)
    float* __restrict__ out)        // (B,20,8)
{
  __shared__ float    sV[A_AA * A_AA];      // 1.6 KB
  __shared__ _Float16 sH0[ROWS * L_SEQ];    // 5 KB
  __shared__ v8h      sW[8 * 16 * 4];       // 8 KB  pool-folded weights
  __shared__ v8h      sB0[ROWS * 258];      // 20.2 KB (lengths 256/64/16/4)
  __shared__ v8h      sB1[ROWS * 130];      // 10.2 KB (lengths 128/32/8/2)

  const int tid  = threadIdx.x;
  const int lane = tid & 31;
  const int wave = tid >> 5;
  const int b    = blockIdx.x >> 2;           // sequence index
  const int k0   = (blockIdx.x & 3) * ROWS;   // first amino row of this WG
  const int rb   = b * A_AA + k0;

  // ---- V = clip(lpm,1e-3,1)*pm, symmetrized, diag=0, col 19=0 ----
  for (int idx = tid; idx < A_AA * A_AA; idx += WGS) {
    int i = idx / A_AA;
    int k = idx - i * A_AA;
    float v = 0.0f;
    if (k != i && k != A_AA - 1) {
      int lo = i < k ? i : k;
      int hh = i < k ? k : i;
      float lv = fminf(fmaxf(lpm[lo * A_AA + hh], 1e-3f), 1.0f);
      v = lv * pm[lo * A_AA + hh];
    }
    sV[idx] = v;
  }

  // ---- pool-folded fp16 weights: sW[l][m][tau][c], rows m=8..15 zeroed ----
  {
    _Float16* sWh = (_Float16*)sW;
    for (int idx = tid; idx < 8 * 8 * 8; idx += WGS) {   // (l, m, c)
      int l = idx >> 6, mm = (idx >> 3) & 7, c = idx & 7;
      const float* w = wr + l * 192 + mm * 24 + c * 3;
      float w0 = w[0], w1 = w[1], w2 = w[2];
      _Float16* dst = sWh + ((l * 16 + mm) * 4) * 8 + c;
      dst[0]  = (_Float16)(0.5f * w0);
      dst[8]  = (_Float16)(0.5f * (w0 + w1));
      dst[16] = (_Float16)(0.5f * (w1 + w2));
      dst[24] = (_Float16)(0.5f * w2);
    }
    for (int idx = tid; idx < 8 * 8 * 4; idx += WGS) {   // (l, m-8, tau)
      int l = idx >> 5, mm = 8 + ((idx >> 2) & 7), tau = idx & 3;
      sW[(l * 16 + mm) * 4 + tau] = (v8h){};
    }
  }
  __syncthreads();

  // ---- h0[r][p] = x[b,p,k0+r] + sum_i x[b,p,i]*V[i,k0+r]  (x is exactly 0/1;
  //      the Gaussian "row" term underflows to 0 in fp32) ----
  for (int p = tid; p < L_SEQ; p += WGS) {
    const float* xr = x + ((size_t)b * L_SEQ + p) * A_AA;   // 80B => 16B aligned
    float xv[20];
#pragma unroll
    for (int w = 0; w < 5; ++w) {
      v4f t = ((const v4f*)xr)[w];
      xv[4 * w + 0] = t[0]; xv[4 * w + 1] = t[1];
      xv[4 * w + 2] = t[2]; xv[4 * w + 3] = t[3];
    }
    float acc[ROWS];
#pragma unroll
    for (int r = 0; r < ROWS; ++r) acc[r] = xv[k0 + r];
#pragma unroll
    for (int i = 0; i < A_AA; ++i) {
#pragma unroll
      for (int r = 0; r < ROWS; ++r) acc[r] += xv[i] * sV[i * A_AA + k0 + r];
    }
#pragma unroll
    for (int r = 0; r < ROWS; ++r) sH0[r * L_SEQ + p] = (_Float16)acc[r];
  }
  __syncthreads();

  // ---- layer 0: conv(1->8,k=3,pad1)+pool, scalar, pool-folded taps ----
  float w0t[24];
#pragma unroll
  for (int j = 0; j < 24; ++j) w0t[j] = wf[j];

  for (int idx = tid; idx < ROWS * 256; idx += WGS) {
    int r  = idx >> 8;
    int q  = idx & 255;
    int p0 = q << 1;
    const _Float16* h = sH0 + r * L_SEQ;
    float v0 = (p0 >= 1) ? (float)h[p0 - 1] : 0.0f;
    float v1 = (float)h[p0];
    float v2 = (float)h[p0 + 1];
    float v3 = (p0 + 2 < L_SEQ) ? (float)h[p0 + 2] : 0.0f;
    v8h pk;
#pragma unroll
    for (int o = 0; o < C_CH; ++o) {
      float s = w0t[o * 3 + 0] * (v0 + v1) + w0t[o * 3 + 1] * (v1 + v2) +
                w0t[o * 3 + 2] * (v2 + v3);
      pk[o] = (_Float16)(0.5f * s);
    }
    sB0[r * 258 + q + 1] = pk;                 // channel-interleaved slot
  }
  if (tid < ROWS * 2) {                        // zero pads: slots 0 and 257
    int r = tid >> 1;
    sB0[r * 258 + ((tid & 1) ? 257 : 0)] = (v8h){};
  }
  __syncthreads();

  // ---- layers 1..8: fused conv+pool WMMA, ping-pong LDS ----
  conv_pool_layer<256, 0, false>(sB0, sB1, sW, out, lane, wave, rb);
  conv_pool_layer<128, 1, false>(sB1, sB0, sW, out, lane, wave, rb);
  conv_pool_layer<64,  2, false>(sB0, sB1, sW, out, lane, wave, rb);
  conv_pool_layer<32,  3, false>(sB1, sB0, sW, out, lane, wave, rb);
  conv_pool_layer<16,  4, false>(sB0, sB1, sW, out, lane, wave, rb);
  conv_pool_layer<8,   5, false>(sB1, sB0, sW, out, lane, wave, rb);
  conv_pool_layer<4,   6, false>(sB0, sB1, sW, out, lane, wave, rb);
  conv_pool_layer<2,   7, true >(sB1, sB0, sW, out, lane, wave, rb);
}

extern "C" void kernel_launch(void* const* d_in, const int* in_sizes, int n_in,
                              void* d_out, int out_size, void* d_ws, size_t ws_size,
                              hipStream_t stream) {
  (void)in_sizes; (void)n_in; (void)out_size; (void)d_ws; (void)ws_size;
  const float* x   = (const float*)d_in[0];
  // d_in[1] = masks (unused by the forward pass)
  const float* lpm = (const float*)d_in[2];
  const float* pm  = (const float*)d_in[3];
  const float* wf  = (const float*)d_in[4];
  const float* wr  = (const float*)d_in[5];
  float* outp      = (float*)d_out;

  dim3 grid(B_SEQ * (A_AA / ROWS));   // 2048 blocks
  dim3 block(WGS);
  hipLaunchKernelGGL(protcnn_fused, grid, block, 0, stream,
                     x, lpm, pm, wf, wr, outp);
}
